// _FlatLinear_23802708754446
// MI455X (gfx1250) — compile-verified, matrix operand order checked by reading, not run on hardware
//
#include <hip/hip_runtime.h>
#include <hip/hip_bf16.h>

typedef __attribute__((ext_vector_type(16))) _Float16 v16h;
typedef __attribute__((ext_vector_type(8)))  _Float16 v8h;   // 16 bytes
typedef __attribute__((ext_vector_type(8)))  float    v8f;

// Problem dims (fixed by the reference)
#define MDIM 8192   // 4 * 2048
#define NDIM 4096   // out_features
#define KDIM 4096   // in_features

// ---------------------------------------------------------------------------
// Kernel 1: convert x (fp32) -> f16, 8 elements per thread (b128 in, b128 out)
// ---------------------------------------------------------------------------
__global__ void cvt_x_kernel(const float* __restrict__ x, _Float16* __restrict__ xh) {
    int i = blockIdx.x * blockDim.x + threadIdx.x;      // i indexes groups of 8
    const float4* x4 = (const float4*)x;
    float4 a = x4[2 * i];
    float4 b = x4[2 * i + 1];
    v8h h;
    h[0] = (_Float16)a.x; h[1] = (_Float16)a.y;
    h[2] = (_Float16)a.z; h[3] = (_Float16)a.w;
    h[4] = (_Float16)b.x; h[5] = (_Float16)b.y;
    h[6] = (_Float16)b.z; h[7] = (_Float16)b.w;
    ((v8h*)xh)[i] = h;
}

// ---------------------------------------------------------------------------
// Kernel 2: dequantize W: wh[i] = (f16) centroids[labels[i]], 8 per thread
// ---------------------------------------------------------------------------
__global__ void dequant_w_kernel(const int* __restrict__ labels,
                                 const float* __restrict__ centroids,
                                 _Float16* __restrict__ wh) {
    int i = blockIdx.x * blockDim.x + threadIdx.x;      // groups of 8
    const int4* l4 = (const int4*)labels;
    int4 l0 = l4[2 * i];
    int4 l1 = l4[2 * i + 1];
    v8h h;
    h[0] = (_Float16)centroids[l0.x]; h[1] = (_Float16)centroids[l0.y];
    h[2] = (_Float16)centroids[l0.z]; h[3] = (_Float16)centroids[l0.w];
    h[4] = (_Float16)centroids[l1.x]; h[5] = (_Float16)centroids[l1.y];
    h[6] = (_Float16)centroids[l1.z]; h[7] = (_Float16)centroids[l1.w];
    ((v8h*)wh)[i] = h;
}

// ---------------------------------------------------------------------------
// Kernel 3: WMMA GEMM  out[m,n] = sum_k xh[m,k] * wh[n,k] + bias[n]
//   Block: 256 threads = 8 waves (2 M-waves x 4 N-waves)
//   Block tile: 128(M) x 256(N); wave tile 64x64 = 4x4 frags of 16x16
//   K-step: 32 per v_wmma_f32_16x16x32_f16
//
// 16-bit A/B fragment layout (ISA 7.12.2), per lane:
//   lanes 0-15  (row = lane):    VGPR0-3 = K 0..7,  VGPR4-7 = K 16..23
//   lanes 16-31 (row = lane-16): VGPR0-3 = K 8..15, VGPR4-7 = K 24..31
// -> two 16B loads per fragment at byte offsets (half*16) and (half*16 + 32).
// ---------------------------------------------------------------------------
__global__ __launch_bounds__(256, 1)
void wmma_gemm_kernel(const _Float16* __restrict__ xh,
                      const _Float16* __restrict__ wh,
                      const float* __restrict__ bias,
                      float* __restrict__ out) {
    const int lane  = threadIdx.x & 31;
    const int wave  = threadIdx.x >> 5;
    const int waveM = wave >> 2;            // 0..1
    const int waveN = wave & 3;             // 0..3
    const int lrow  = lane & 15;            // row within 16x16 frag
    const int lhalf = lane >> 4;            // 0 or 1 -> which K half

    const int m0 = blockIdx.y * 128 + waveM * 64;
    const int n0 = blockIdx.x * 256 + waveN * 64;

    v8f acc[4][4];
#pragma unroll
    for (int i = 0; i < 4; ++i)
#pragma unroll
        for (int j = 0; j < 4; ++j)
            acc[i][j] = (v8f){};

    for (int k0 = 0; k0 < KDIM; k0 += 32) {
        v16h afrag[4], bfrag[4];

#pragma unroll
        for (int im = 0; im < 4; ++im) {
            const _Float16* p =
                xh + (size_t)(m0 + im * 16 + lrow) * KDIM + k0 + lhalf * 8;
            v8h lo = *(const v8h*)p;          // K  0..7  (or 8..15)
            v8h hi = *(const v8h*)(p + 16);   // K 16..23 (or 24..31)
#pragma unroll
            for (int j = 0; j < 8; ++j) {
                afrag[im][j]     = lo[j];
                afrag[im][j + 8] = hi[j];
            }
        }
#pragma unroll
        for (int in = 0; in < 4; ++in) {
            const _Float16* p =
                wh + (size_t)(n0 + in * 16 + lrow) * KDIM + k0 + lhalf * 8;
            v8h lo = *(const v8h*)p;
            v8h hi = *(const v8h*)(p + 16);
#pragma unroll
            for (int j = 0; j < 8; ++j) {
                bfrag[in][j]     = lo[j];
                bfrag[in][j + 8] = hi[j];
            }
        }

#pragma unroll
        for (int im = 0; im < 4; ++im)
#pragma unroll
            for (int in = 0; in < 4; ++in)
                acc[im][in] = __builtin_amdgcn_wmma_f32_16x16x32_f16(
                    /*neg_a=*/false, afrag[im],
                    /*neg_b=*/false, bfrag[in],
                    /*c_mod=*/(short)0, acc[im][in],
                    /*reuse_a=*/false, /*reuse_b=*/false);
    }

    // Epilogue: C layout -> VGPR v holds M = v + 8*lhalf, column N = lrow.
#pragma unroll
    for (int in = 0; in < 4; ++in) {
        const int n  = n0 + in * 16 + lrow;
        const float bv = bias[n];
#pragma unroll
        for (int im = 0; im < 4; ++im) {
            const int mbase = m0 + im * 16 + lhalf * 8;
#pragma unroll
            for (int v = 0; v < 8; ++v) {
                out[(size_t)(mbase + v) * NDIM + n] = acc[im][in][v] + bv;
            }
        }
    }
}

// ---------------------------------------------------------------------------
// Launcher
//   d_in[0]=x (fp32, 8192*4096), d_in[1]=centroids (fp32, 256),
//   d_in[2]=labels (int32, 4096*4096), d_in[3]=bias (fp32, 4096)
//   d_out = fp32, 8192*4096
//   d_ws: [0, 64MiB) = xh (f16), [64MiB, 96MiB) = wh (f16)
// ---------------------------------------------------------------------------
extern "C" void kernel_launch(void* const* d_in, const int* in_sizes, int n_in,
                              void* d_out, int out_size, void* d_ws, size_t ws_size,
                              hipStream_t stream) {
    const float* x         = (const float*)d_in[0];
    const float* centroids = (const float*)d_in[1];
    const int*   labels    = (const int*)d_in[2];
    const float* bias      = (const float*)d_in[3];
    float*       out       = (float*)d_out;

    _Float16* xh = (_Float16*)d_ws;
    _Float16* wh = (_Float16*)((char*)d_ws + (size_t)MDIM * KDIM * sizeof(_Float16));

    // x: 8192*4096 = 33,554,432 elems -> /8 per thread -> 4,194,304 threads
    cvt_x_kernel<<<(MDIM * KDIM) / (8 * 256), 256, 0, stream>>>(x, xh);
    // W: 4096*4096 = 16,777,216 elems -> 2,097,152 threads
    dequant_w_kernel<<<(NDIM * KDIM) / (8 * 256), 256, 0, stream>>>(labels, centroids, wh);

    dim3 grid(NDIM / 256, MDIM / 128);   // (16, 64)
    wmma_gemm_kernel<<<grid, 256, 0, stream>>>(xh, wh, bias, out);
}
